// GRU_70480413327664
// MI455X (gfx1250) — matrix operand order repaired and checked by reading, not compiled, not run on hardware
//
#include <hip/hip_runtime.h>

typedef __attribute__((ext_vector_type(16))) _Float16 v16h;
typedef __attribute__((ext_vector_type(8)))  _Float16 v8h;
typedef __attribute__((ext_vector_type(8)))  float    v8f;
typedef __attribute__((ext_vector_type(4)))  float    v4f;

#define HIDDEN  64
#define CLASSES 5
#define ROWS    16      // batch rows per block
#define CHUNK   256     // x timesteps staged in LDS per refill
#define HPAD    72      // 64 + 8 halves padding (144B row stride, 16B aligned)

#define NLOG2E  (-1.44269504088896340736f)
#define LOG2E2  2.88539008177792681472f

__device__ __forceinline__ float fast_rcp(float x) { return __builtin_amdgcn_rcpf(x); }
__device__ __forceinline__ float exp2_hw(float x)  { return __builtin_amdgcn_exp2f(x); }

// sigmoid with argument pre-scaled by -log2(e): sigmoid = rcp(1 + exp2(arg2))
__device__ __forceinline__ float sigmoid_pre(float arg2) {
    return fast_rcp(1.0f + exp2_hw(arg2));   // exp2->inf => rcp(inf)=0, safe
}
__device__ __forceinline__ float fast_tanh(float x) {
    // tanh(|x|) = 1 - 2/(2^{2log2e*|x|}+1); exp2->inf => rcp->0 => t=1, safe
    float e = exp2_hw(LOG2E2 * fabsf(x));    // abs is a free src modifier
    float t = 1.0f - 2.0f * fast_rcp(e + 1.0f);
    return (x < 0.0f) ? -t : t;
}

__global__ __launch_bounds__(128) void gru_wmma_kernel(
    const float* __restrict__ x,      // [B, T, 1]
    const float* __restrict__ w_ih,   // [3H, 1]
    const float* __restrict__ w_hh,   // [3H, H]
    const float* __restrict__ b_ih,   // [3H]
    const float* __restrict__ b_hh,   // [3H]
    const float* __restrict__ w_head, // [C, H]
    const float* __restrict__ b_head, // [C]
    float* __restrict__ out,          // [B, C]
    int T)
{
    __shared__ _Float16 hbuf[2][ROWS][HPAD];     // double-buffered h_t (f16, A image)
    __shared__ float    xbuf[CHUNK][ROWS];       // staged x chunk, time-major
    __shared__ float    hfin[ROWS][HIDDEN + 1];  // final h (f32) for head

    const int tid  = threadIdx.x;
    const int lane = tid & 31;
    const int wave = tid >> 5;       // 0..3
    const int hi   = lane >> 4;      // lane half (0/1)
    const int ln   = lane & 15;
    const int rowbase = blockIdx.x * ROWS;

    // Hidden column owned by this lane (C-layout N index, global 0..63)
    const int j = wave * 16 + ln;

    // Per-lane scalars; r/z gate constants pre-scaled by -log2(e) so the
    // sigmoid argument feeds v_exp_f32 (exp2) directly with no extra mul.
    const float wir = NLOG2E * w_ih[j];
    const float wiz = NLOG2E * w_ih[HIDDEN + j];
    const float win = w_ih[2 * HIDDEN + j];
    const float bir = NLOG2E * b_ih[j];
    const float biz = NLOG2E * b_ih[HIDDEN + j];
    const float bin_ = b_ih[2 * HIDDEN + j];
    const float bhr = NLOG2E * b_hh[j];
    const float bhz = NLOG2E * b_hh[HIDDEN + j];
    const float bhn = b_hh[2 * HIDDEN + j];

    // ---- Build resident B-fragments of W_hh^T (f16), 6 fragments per wave ----
    // B 32x16 f16 layout: lane ln -> N = gate_col; element e<8 -> K = 32*c + 8*hi + e, e>=8 -> K = 32*c + 16 + 8*hi + (e-8)
    // B[k][n] = scale * W_hh[g][k], g = gsel*64 + j (row-major, contiguous reads).
    // r/z fragments folded with -log2(e); n fragment unscaled.
    v16h Bf[3][2];
#pragma unroll
    for (int gsel = 0; gsel < 3; ++gsel) {
        const int g = gsel * HIDDEN + j;
        const float scale = (gsel < 2) ? NLOG2E : 1.0f;
#pragma unroll
        for (int c = 0; c < 2; ++c) {
            const float* src = w_hh + (size_t)g * HIDDEN + 32 * c + 8 * hi;
            v16h b;
#pragma unroll
            for (int e = 0; e < 16; ++e) b[e] = (_Float16)(scale * src[e + (e >= 8 ? 8 : 0)]);
            Bf[gsel][c] = b;
        }
    }

    // ---- h0 = 0 (buffer 0 is the first read buffer) ----
    for (int i = tid; i < ROWS * HPAD; i += 128) ((_Float16*)hbuf[0])[i] = (_Float16)0.0f;
    v8f hprev;
#pragma unroll
    for (int v = 0; v < 8; ++v) hprev[v] = 0.0f;
    __syncthreads();

    // One GRU timestep: read h_t from hrd, write h_{t+1} to hwr, one barrier.
    auto step = [&](int t, const _Float16 (*hrd)[HPAD], _Float16 (*hwr)[HPAD]) {
        // ---- load A-fragments (h_t, 16x64 f16) from LDS ----
        // A 16x32 f16 layout: lane -> row = ln; elems 0..7 -> K = 32c + 8*hi + e,
        // elems 8..15 -> K = 32c + 16 + 8*hi + (e-8): two contiguous 16B runs.
        v16h A0, A1;
        {
            const _Float16* base = &hrd[ln][8 * hi];
            v8h l0 = *(const v8h*)(base);
            v8h h0 = *(const v8h*)(base + 16);
            v8h l1 = *(const v8h*)(base + 32);
            v8h h1 = *(const v8h*)(base + 48);
#pragma unroll
            for (int e = 0; e < 8; ++e) {
                A0[e] = l0[e]; A0[8 + e] = h0[e];
                A1[e] = l1[e]; A1[8 + e] = h1[e];
            }
        }
        // per-step x for this lane's 8 rows: contiguous -> 2x ds_load_b128
        const int tc = t & (CHUNK - 1);
        v4f xlo = *(const v4f*)&xbuf[tc][8 * hi];
        v4f xhi = *(const v4f*)&xbuf[tc][8 * hi + 4];

        // ---- gh = h @ W_hh^T (+ b_hh folded into C-init; r/z pre-scaled) ----
        v8f accR, accZ, accN;
#pragma unroll
        for (int v = 0; v < 8; ++v) { accR[v] = bhr; accZ[v] = bhz; accN[v] = bhn; }

        accR = __builtin_amdgcn_wmma_f32_16x16x32_f16(false, A0, false, Bf[0][0], (short)0, accR, false, false);
        accR = __builtin_amdgcn_wmma_f32_16x16x32_f16(false, A1, false, Bf[0][1], (short)0, accR, false, false);
        accZ = __builtin_amdgcn_wmma_f32_16x16x32_f16(false, A0, false, Bf[1][0], (short)0, accZ, false, false);
        accZ = __builtin_amdgcn_wmma_f32_16x16x32_f16(false, A1, false, Bf[1][1], (short)0, accZ, false, false);
        accN = __builtin_amdgcn_wmma_f32_16x16x32_f16(false, A0, false, Bf[2][0], (short)0, accN, false, false);
        accN = __builtin_amdgcn_wmma_f32_16x16x32_f16(false, A1, false, Bf[2][1], (short)0, accN, false, false);

        // ---- gates + state update (in-lane; C layout: row m = v + 8*hi, col j) ----
        v8f hnew;
#pragma unroll
        for (int v = 0; v < 8; ++v) {
            float xv = (v < 4) ? xlo[v & 3] : xhi[v & 3];
            float r  = sigmoid_pre(fmaf(xv, wir, bir) + accR[v]);
            float z  = sigmoid_pre(fmaf(xv, wiz, biz) + accZ[v]);
            float n  = fast_tanh(fmaf(xv, win, bin_) + r * accN[v]);
            hnew[v]  = n + z * (hprev[v] - n);   // == (1-z)*n + z*h
        }
        hprev = hnew;

        // ---- publish h_{t+1} as f16 into the other buffer ----
#pragma unroll
        for (int v = 0; v < 8; ++v) hwr[v + 8 * hi][j] = (_Float16)hnew[v];
        __syncthreads();   // single barrier per step (double-buffered h)
    };

    auto refill = [&](int t) {
        for (int i = tid; i < ROWS * CHUNK; i += 128) {
            int r  = i / CHUNK;           // coalesced global read
            int cc = i % CHUNK;
            int tt = t + cc;
            xbuf[cc][r] = (tt < T) ? x[(size_t)(rowbase + r) * T + tt] : 0.0f;
        }
        if (t + CHUNK < T) {
            // 16 rows x 8 groups x 128B = full next chunk (global_prefetch_b8)
            const float* nb = x + (size_t)(rowbase + (tid & 15)) * T + (t + CHUNK)
                              + ((tid >> 4) << 5);
            __builtin_prefetch(nb, 0, 3);
        }
        __syncthreads();
    };

    // Unrolled by 2: buffer parities are compile-time, no per-step bookkeeping.
    // CHUNK is even, so refills only ever trigger on the even step.
    const int Teven = T & ~1;
    for (int t = 0; t < Teven; t += 2) {
        if ((t & (CHUNK - 1)) == 0) refill(t);
        step(t,     hbuf[0], hbuf[1]);
        step(t + 1, hbuf[1], hbuf[0]);
    }
    if (T & 1) {
        if ((Teven & (CHUNK - 1)) == 0) refill(Teven);
        step(Teven, hbuf[0], hbuf[1]);
    }

    // ---- head: out = h_last @ w_head^T + b_head ----
#pragma unroll
    for (int v = 0; v < 8; ++v) hfin[v + 8 * hi][j] = hprev[v];
    __syncthreads();

    if (tid < ROWS * CLASSES) {
        int r = tid & 15;
        int c = tid >> 4;   // 0..4
        float s = b_head[c];
#pragma unroll
        for (int k = 0; k < HIDDEN; ++k) s += hfin[r][k] * w_head[c * HIDDEN + k];
        out[(size_t)(rowbase + r) * CLASSES + c] = s;
    }
}

extern "C" void kernel_launch(void* const* d_in, const int* in_sizes, int n_in,
                              void* d_out, int out_size, void* d_ws, size_t ws_size,
                              hipStream_t stream) {
    const float* x      = (const float*)d_in[0];
    const float* w_ih   = (const float*)d_in[1];
    const float* w_hh   = (const float*)d_in[2];
    const float* b_ih   = (const float*)d_in[3];
    const float* b_hh   = (const float*)d_in[4];
    const float* w_head = (const float*)d_in[5];
    const float* b_head = (const float*)d_in[6];
    float* out = (float*)d_out;

    const int B = out_size / CLASSES;      // 512
    const int T = in_sizes[0] / B;         // 3600
    const int nblocks = B / ROWS;          // 32

    gru_wmma_kernel<<<nblocks, 128, 0, stream>>>(x, w_ih, w_hh, b_ih, b_hh,
                                                 w_head, b_head, out, T);
}
